// SimpleDH_SNN_86474871538204
// MI455X (gfx1250) — compile-verified, matrix operand order checked by reading, not run on hardware
//
#include <hip/hip_runtime.h>
#include <hip/hip_bf16.h>
#include <stddef.h>
#include <stdint.h>

// Problem constants (from reference)
#define B_  32
#define T_  1024
#define IN_ 512
#define H_  512
#define O_  256
#define NB_ 4
#define BS_ 128
#define M_  (B_ * T_)   // 32768 rows

typedef __attribute__((ext_vector_type(16))) _Float16 v16h;
typedef __attribute__((ext_vector_type(8)))  _Float16 v8h;
typedef __attribute__((ext_vector_type(4)))  _Float16 v4h;
typedef __attribute__((ext_vector_type(8)))  float    v8f;
typedef __attribute__((ext_vector_type(4)))  float    v4f;

// ---------------------------------------------------------------------------
// Workspace layout (bytes, all 256-aligned)
//   xh   : f16 [M, IN]            33,554,432
//   WbT  : f16 [NB, H, BS]           524,288
//   WoT  : f16 [O, H]                262,144
//   beta : f32 [NB, H]                 8,192
//   alpha: f32 [H]                     2,048
//   din  : f16 [M, NB, H]        134,217,728   (fits in 192 MB L2 between k1/k2)
//   memh : f16 [M, H]             33,554,432
// ---------------------------------------------------------------------------
#define OFF_XH    ((size_t)0)
#define OFF_WBT   ((size_t)33554432)
#define OFF_WOT   ((size_t)34078720)
#define OFF_BETA  ((size_t)34340864)
#define OFF_ALPHA ((size_t)34349056)
#define OFF_DIN   ((size_t)34351104)
#define OFF_MEM   ((size_t)168568832)

// Load a 16-half WMMA operand fragment (A-type or B-type; both use the same
// per-lane K packing for v_wmma_f32_16x16x32_f16):
//   lanes 0-15 : K = {k0+0..7,  k0+16..23}
//   lanes 16-31: K = {k0+8..15, k0+24..31}
// `rowbase` must already point at element [row, k0 + half*8] of a row-major
// matrix (16-byte aligned). Works for both global and LDS pointers.
__device__ __forceinline__ v16h load_frag(const _Float16* rowbase) {
    v8h lo = *(const v8h*)(rowbase);
    v8h hi = *(const v8h*)(rowbase + 16);
    return __builtin_shufflevector(lo, hi, 0, 1, 2, 3, 4, 5, 6, 7,
                                           8, 9, 10, 11, 12, 13, 14, 15);
}

// ---------------------------------------------------------------------------
// Kernel 0a: x f32 -> f16 (vectorized, 4 elems/thread, exact coverage).
// x is read exactly once -> nontemporal load keeps L2 free for din/memh.
// ---------------------------------------------------------------------------
__global__ __launch_bounds__(256) void prep_x_kernel(const float* __restrict__ x,
                                                     _Float16* __restrict__ xh) {
    int i = (blockIdx.x * 256 + threadIdx.x) * 4;
    v4f v = __builtin_nontemporal_load((const v4f*)(x + i));
    v4h o = {(_Float16)v.x, (_Float16)v.y, (_Float16)v.z, (_Float16)v.w};
    *(v4h*)(xh + i) = o;
}

// ---------------------------------------------------------------------------
// Kernel 0b: weight transposes + gate precompute
//   WbT[n][h][c] = Wb[n][c][h]   (f16)
//   WoT[o][h]    = Wo[h][o]      (f16)
//   beta[n][h]   = sigmoid(tau_n[h][n])
//   alpha[h]     = sigmoid(tau_m[h])
// ---------------------------------------------------------------------------
__global__ __launch_bounds__(256) void prep_w_kernel(const float* __restrict__ Wb,
                                                     const float* __restrict__ Wo,
                                                     const float* __restrict__ tau_m,
                                                     const float* __restrict__ tau_n,
                                                     _Float16* __restrict__ WbT,
                                                     _Float16* __restrict__ WoT,
                                                     float* __restrict__ beta,
                                                     float* __restrict__ alpha) {
    int id = blockIdx.x * 256 + threadIdx.x;
    const int nWbT = NB_ * H_ * BS_;   // 262144
    const int nWoT = O_ * H_;          // 131072
    const int nBeta = NB_ * H_;        // 2048
    if (id < nWbT) {
        int n = id / (H_ * BS_);
        int r = id - n * (H_ * BS_);
        int h = r / BS_;
        int c = r - h * BS_;
        WbT[id] = (_Float16)Wb[(n * BS_ + c) * H_ + h];
        return;
    }
    id -= nWbT;
    if (id < nWoT) {
        int o = id / H_;
        int h = id - o * H_;
        WoT[o * H_ + h] = (_Float16)Wo[h * O_ + o];
        return;
    }
    id -= nWoT;
    if (id < nBeta) {
        int n = id / H_;
        int h = id - n * H_;
        beta[id] = 1.0f / (1.0f + __expf(-tau_n[h * NB_ + n]));
        return;
    }
    id -= nBeta;
    if (id < H_) {
        alpha[id] = 1.0f / (1.0f + __expf(-tau_m[id]));
    }
}

// ---------------------------------------------------------------------------
// Kernel 1: branch GEMM via WMMA.
//   din[m, n, h] = sum_c xh[m, n*BS + c] * WbT[n, h, c] + bb[n, h]
// blockIdx.x encodes (mtile, n); 8 waves cover H=512 in 16x64 tiles.
// A strip (16x128) is fully preloaded (4 fragments); B fragments are
// double-buffered so k+1 loads overlap k's WMMAs. Results are staged in
// wave-private LDS and written as coalesced 16-byte stores (din is the
// dominant 134 MB stream).
// ---------------------------------------------------------------------------
__global__ __launch_bounds__(256) void gemm1_kernel(const _Float16* __restrict__ xh,
                                                    const _Float16* __restrict__ WbT,
                                                    const float* __restrict__ bb,
                                                    _Float16* __restrict__ din) {
    __shared__ __align__(16) _Float16 stage[8][16 * 64];   // 2 KB per wave

    const int lane = threadIdx.x & 31;
    const int wave = threadIdx.x >> 5;
    const int mtile = blockIdx.x >> 2;
    const int n = blockIdx.x & 3;
    const int m0 = mtile * 16;
    const int h0 = wave * 64;
    const int col = lane & 15;
    const int half = lane >> 4;

    const _Float16* arow = xh + (size_t)(m0 + col) * IN_ + n * BS_ + half * 8;
    const _Float16* bbase = WbT + (size_t)n * (H_ * BS_);
    const _Float16* brow[4];
#pragma unroll
    for (int j = 0; j < 4; ++j)
        brow[j] = bbase + (size_t)(h0 + j * 16 + col) * BS_ + half * 8;

    // full K=128 A strip: 4 fragments (32 VGPRs)
    v16h a[4];
#pragma unroll
    for (int kk = 0; kk < 4; ++kk) a[kk] = load_frag(arow + kk * 32);

    v8f acc[4] = {{}, {}, {}, {}};
    v16h bfr[2][4];
#pragma unroll
    for (int j = 0; j < 4; ++j) bfr[0][j] = load_frag(brow[j]);

#pragma unroll
    for (int kk = 0; kk < 4; ++kk) {
        const int cur = kk & 1;
        if (kk < 3) {
#pragma unroll
            for (int j = 0; j < 4; ++j)
                bfr[cur ^ 1][j] = load_frag(brow[j] + (kk + 1) * 32);
        }
#pragma unroll
        for (int j = 0; j < 4; ++j)
            acc[j] = __builtin_amdgcn_wmma_f32_16x16x32_f16(
                false, a[kk], false, bfr[cur][j], (short)0, acc[j], false, false);
    }

    // C/D layout: VGPR i -> (row = half*8 + i, col = lane&15).
    // Stage row-major 16x64 tile in LDS (same-wave DS ordering, no barrier),
    // then 4 coalesced b128 stores per lane.
    _Float16* st = &stage[wave][0];
#pragma unroll
    for (int j = 0; j < 4; ++j) {
        const float bbv = bb[n * H_ + h0 + j * 16 + col];
#pragma unroll
        for (int i = 0; i < 8; ++i)
            st[(half * 8 + i) * 64 + j * 16 + col] = (_Float16)(acc[j][i] + bbv);
    }
#pragma unroll
    for (int c = 0; c < 4; ++c) {
        const int chunk = c * 32 + lane;      // 128 chunks of 8 halfs
        const int r = chunk >> 3;
        const int cc = (chunk & 7) * 8;
        v8h vv = *(const v8h*)(st + r * 64 + cc);
        *(v8h*)(din + ((size_t)(m0 + r) * NB_ + n) * H_ + h0 + cc) = vv;
    }
}

// ---------------------------------------------------------------------------
// Kernel 2: the only sequential part — per-(b,h) leaky-integrator scan over T.
// One thread per (b,h): 16384 threads = 512 wave32 waves, pure streaming.
// ---------------------------------------------------------------------------
__global__ __launch_bounds__(256) void scan_kernel(const _Float16* __restrict__ din,
                                                   const float* __restrict__ beta,
                                                   const float* __restrict__ alpha,
                                                   _Float16* __restrict__ memh) {
    const int idx = blockIdx.x * 256 + threadIdx.x;   // 0 .. B*H-1
    const int b = idx / H_;
    const int h = idx - b * H_;

    float be[NB_], ombe[NB_], dc[NB_];
#pragma unroll
    for (int n = 0; n < NB_; ++n) {
        be[n] = beta[n * H_ + h];
        ombe[n] = 1.0f - be[n];
        dc[n] = 0.0f;
    }
    const float al = alpha[h];
    const float omal = 1.0f - al;
    float mem = 0.0f;

    const _Float16* p = din + (size_t)b * T_ * NB_ * H_ + h;
    _Float16* q = memh + (size_t)b * T_ * H_ + h;
    for (int t = 0; t < T_; ++t) {
        float tot = 0.0f;
#pragma unroll
        for (int n = 0; n < NB_; ++n) {
            const float di = (float)p[n * H_];
            dc[n] = be[n] * dc[n] + ombe[n] * di;
            tot += dc[n];
        }
        mem = al * mem + omal * tot;
        *q = (_Float16)mem;
        p += NB_ * H_;
        q += H_;
    }
}

// ---------------------------------------------------------------------------
// Kernel 3: output projection as one big WMMA GEMM + fused bias/sigmoid.
//   out[m, o] = sigmoid( sum_h memh[m,h] * WoT[o,h] + bo[o] )
// A tile (16 x 512 f16, 16 KB) staged straight into LDS with gfx1250 async
// loads (ASYNCcnt, bypasses VGPRs); 8 waves each compute a 16x32 slice of
// O=256 with double-buffered B fragments.
// ---------------------------------------------------------------------------
__global__ __launch_bounds__(256) void gemm2_kernel(const _Float16* __restrict__ memh,
                                                    const _Float16* __restrict__ WoT,
                                                    const float* __restrict__ bo,
                                                    float* __restrict__ out) {
    __shared__ __align__(16) _Float16 As[16 * H_];   // 16 KB

    const int m0 = blockIdx.x * 16;
    // async global->LDS staging: 1024 b128 transfers, 4 per thread
    for (int idx = threadIdx.x; idx < (16 * H_) / 8; idx += 256) {
        const int r = idx >> 6;              // H/8 = 64 chunks per row
        const int c = (idx & 63) * 8;
        const _Float16* g = memh + (size_t)(m0 + r) * H_ + c;
        unsigned lds = (unsigned)(uintptr_t)(As + r * H_ + c);
        asm volatile("global_load_async_to_lds_b128 %0, %1, off"
                     :: "v"(lds), "v"(g) : "memory");
    }
    asm volatile("s_wait_asynccnt 0x0" ::: "memory");
    __syncthreads();

    const int lane = threadIdx.x & 31;
    const int wave = threadIdx.x >> 5;
    const int col = lane & 15;
    const int half = lane >> 4;
    const int o0 = wave * 32;

    const _Float16* arow = As + col * H_ + half * 8;              // LDS fragments
    const _Float16* brow0 = WoT + (size_t)(o0 + col) * H_ + half * 8;
    const _Float16* brow1 = brow0 + (size_t)16 * H_;

    v8f acc0 = {}, acc1 = {};
    v16h b0[2], b1[2];
    b0[0] = load_frag(brow0);
    b1[0] = load_frag(brow1);

#pragma unroll
    for (int kk = 0; kk < 16; ++kk) {
        const int cur = kk & 1;
        if (kk < 15) {
            __builtin_prefetch(brow0 + (kk + 2) * 32, 0, 3);
            b0[cur ^ 1] = load_frag(brow0 + (kk + 1) * 32);
            b1[cur ^ 1] = load_frag(brow1 + (kk + 1) * 32);
        }
        v16h af = load_frag(arow + kk * 32);
        acc0 = __builtin_amdgcn_wmma_f32_16x16x32_f16(false, af, false, b0[cur],
                                                      (short)0, acc0, false, false);
        acc1 = __builtin_amdgcn_wmma_f32_16x16x32_f16(false, af, false, b1[cur],
                                                      (short)0, acc1, false, false);
    }

    v8f accs[2] = {acc0, acc1};
#pragma unroll
    for (int j = 0; j < 2; ++j) {
        const int o = o0 + j * 16 + col;
        const float bov = bo[o];
#pragma unroll
        for (int i = 0; i < 8; ++i) {
            const int r = m0 + half * 8 + i;
            const float v = accs[j][i] + bov;
            out[(size_t)r * O_ + o] = 1.0f / (1.0f + __expf(-v));
        }
    }
}

// ---------------------------------------------------------------------------
extern "C" void kernel_launch(void* const* d_in, const int* in_sizes, int n_in,
                              void* d_out, int out_size, void* d_ws, size_t ws_size,
                              hipStream_t stream) {
    (void)in_sizes; (void)n_in; (void)out_size; (void)ws_size;

    const float* x     = (const float*)d_in[0];
    const float* Wb    = (const float*)d_in[1];
    const float* bb    = (const float*)d_in[2];
    const float* Wo    = (const float*)d_in[3];
    const float* bo    = (const float*)d_in[4];
    const float* tau_m = (const float*)d_in[5];
    const float* tau_n = (const float*)d_in[6];
    float* out = (float*)d_out;

    char* ws = (char*)d_ws;
    _Float16* xh    = (_Float16*)(ws + OFF_XH);
    _Float16* WbT   = (_Float16*)(ws + OFF_WBT);
    _Float16* WoT   = (_Float16*)(ws + OFF_WOT);
    float*    beta  = (float*)(ws + OFF_BETA);
    float*    alpha = (float*)(ws + OFF_ALPHA);
    _Float16* din   = (_Float16*)(ws + OFF_DIN);
    _Float16* memh  = (_Float16*)(ws + OFF_MEM);

    // 0a) x -> f16                 (16,777,216 elems, 4/thread)
    prep_x_kernel<<<16384, 256, 0, stream>>>(x, xh);
    // 0b) weight transposes + gates (395,776 work items)
    prep_w_kernel<<<1547, 256, 0, stream>>>(Wb, Wo, tau_m, tau_n, WbT, WoT, beta, alpha);
    // 1) branch GEMM: (mtile, n) per block, 8 waves x 16x64 tiles
    gemm1_kernel<<<(M_ / 16) * NB_, 256, 0, stream>>>(xh, WbT, bb, din);
    // 2) temporal scan: one thread per (b, h)
    scan_kernel<<<(B_ * H_) / 256, 256, 0, stream>>>(din, beta, alpha, memh);
    // 3) output GEMM + sigmoid: one block per 16-row tile
    gemm2_kernel<<<M_ / 16, 256, 0, stream>>>(memh, WoT, bo, out);
}
